// TLSTM_1760936592020
// MI455X (gfx1250) — matrix operand order repaired and checked by reading, not compiled, not run on hardware
//
#include <hip/hip_runtime.h>

// ---------------------------------------------------------------------------
// TLSTM for MI455X (gfx1250, wave32, WMMA).
// Recurrence weight channels: Ui -> VGPRs, Uf -> LDS (padded pitch),
// Uo/Uc/Wd -> streamed from L2 with a per-step K-rotation (kk = (kc+s)&7)
// that defeats LICM (previous round showed the compiler hoisting the
// loop-invariant weight loads and spilling them to scratch).
// G stored gate-interleaved [row][4][256] for single-base immediate-offset
// loads. Raw v_exp/v_log/v_rcp for the activations.
// ---------------------------------------------------------------------------

typedef __attribute__((ext_vector_type(16))) __bf16        v16bf;
typedef __attribute__((ext_vector_type(8)))  float         v8f;
typedef __attribute__((ext_vector_type(4)))  unsigned int  u32x4;

union AFrag {            // 8 VGPRs: one 16x32 bf16 A-frag or 32x16 bf16 B-frag
    v16bf v;
    u32x4 q[2];
};

__device__ __forceinline__ v8f wmma_bf16(v16bf a, v16bf b, v8f c) {
    return __builtin_amdgcn_wmma_f32_16x16x32_bf16(
        /*neg_a=*/false, a, /*neg_b=*/false, b,
        /*c_mod=*/(short)0, c, /*reuse_a=*/false, /*reuse_b=*/false);
}

__device__ __forceinline__ float fast_rcp(float x) {
    return __builtin_amdgcn_rcpf(x);                    // v_rcp_f32
}
__device__ __forceinline__ float fast_exp(float x) {    // e^x, range-safe here
    return __builtin_amdgcn_exp2f(x * 1.4426950408889634f);
}
__device__ __forceinline__ float fast_log(float x) {    // ln(x), x in [2.7,13]
    return __builtin_amdgcn_logf(x) * 0.6931471805599453f;
}
__device__ __forceinline__ float sigmoid_f(float x) {
    return fast_rcp(1.0f + fast_exp(-x));
}
__device__ __forceinline__ float tanh_f(float x) {
    float ax = fabsf(x);
    float e  = fast_exp(-2.0f * ax);
    float t  = (1.0f - e) * fast_rcp(1.0f + e);
    return copysignf(t, x);
}

#define NROW 32768        // B*S

// ----------------- Phase 2: persistent recurrence --------------------------
// grid = 4 (batch slices of 16 rows), block = 512 (16 waves).
// Wave w owns N-tile w for all 5 recurrent GEMMs -> cell update is lane-local.
__global__ __launch_bounds__(512, 1) void tlstm_rec_kernel(
    const __bf16* __restrict__ U5,      // [5][256][256] bf16 (Ui,Uf,Uo,Uc,Wd)
    const float*  __restrict__ G,       // [32768][4][256] fp32 (i,f,o,cand)
    const float*  __restrict__ ts,      // [64][512] fp32
    const float* __restrict__ Uib, const float* __restrict__ Ufb,
    const float* __restrict__ Uob, const float* __restrict__ Ucb,
    const float* __restrict__ Wdb, const float* __restrict__ bd_,
    float* __restrict__ out, int writeTriple)
{
    __shared__ __align__(16) __bf16 hbf[16][264];        //  8448 B
    __shared__ __align__(16) __bf16 cbf[16][264];        //  8448 B
    __shared__ __align__(16) __bf16 Wlds[256 * 264];     // Uf, padded pitch

    const int tid   = threadIdx.x;
    const int wave  = tid >> 5;
    const int lane  = tid & 31;
    const int n     = lane & 15;
    const int hi    = lane >> 4;
    const int col   = wave * 16 + n;
    const int slice = blockIdx.x;           // 0..3, rows [16*slice,16*slice+16)

    // zero the h/c bf16 operand copies
    for (int e = tid; e < 16 * 264; e += 512) {
        ((__bf16*)hbf)[e] = (__bf16)0.0f;
        ((__bf16*)cbf)[e] = (__bf16)0.0f;
    }
    // copy Uf (gate 1) into LDS with padded pitch (16B chunks)
    for (int e = tid; e < 8192; e += 512) {
        int row = e >> 5;
        int k8  = (e & 31) * 8;
        *(u32x4*)&Wlds[row * 264 + k8] =
            *(const u32x4*)&U5[65536 + row * 256 + k8];
    }

    // cell state carried in fp32 registers, one element per (lane, r)
    float creg[8];
#pragma unroll
    for (int r = 0; r < 8; ++r) creg[r] = 0.0f;

    // per-lane base indices
    size_t gBase[8];     // into G  [row][4][256]
    size_t oBase[8];     // into out[b][s][256]
    int    tsBase[8];
#pragma unroll
    for (int r = 0; r < 8; ++r) {
        int bglob = slice * 16 + r + 8 * hi;
        gBase[r]  = (size_t)bglob * (512 * 1024) + col;
        oBase[r]  = (size_t)bglob * (512 * 256)  + col;
        tsBase[r] = bglob * 512;
    }

    // gate 0 (Ui) B-fragments live in VGPRs across all 512 steps (64 regs)
    AFrag Bw0[8];
#pragma unroll
    for (int kc = 0; kc < 8; ++kc) {
        const __bf16* Wr = U5 + (size_t)col * 256 + kc * 32 + 16 * hi;
        Bw0[kc].q[0] = *(const u32x4*)Wr;
        Bw0[kc].q[1] = *(const u32x4*)(Wr + 8);
    }

    const float bU0 = Uib[col], bU1 = Ufb[col], bU2 = Uob[col], bU3 = Ucb[col];
    const float bD  = Wdb[col] + bd_[col];
    const size_t NSH = (size_t)64 * 512 * 256;

    __syncthreads();       // state zeroed + Uf staged

    for (int s = 0; s < 512; ++s) {
        // prefetch pre-activations + decay; independent of h/c, so they fly
        // during the WMMA chain below
        float x0[8], x1[8], x2[8], x3[8], td[8];
#pragma unroll
        for (int r = 0; r < 8; ++r) {
            size_t gb = gBase[r] + (size_t)s * 1024;
            x0[r] = G[gb];
            x1[r] = G[gb + 256];
            x2[r] = G[gb + 512];
            x3[r] = G[gb + 768];
            td[r] = fast_rcp(fast_log(ts[tsBase[r] + s] + 2.7183f));
        }

        v8f acc[5] = {{0,0,0,0,0,0,0,0},{0,0,0,0,0,0,0,0},{0,0,0,0,0,0,0,0},
                      {0,0,0,0,0,0,0,0},{0,0,0,0,0,0,0,0}};
#pragma unroll
        for (int kc = 0; kc < 8; ++kc) {
            const int kk = (kc + s) & 7;     // rotation: defeats LICM hoisting

            AFrag ah, ahk, ack, b1;
            ah.q[0]  = *(const u32x4*)&hbf[n][kc * 32 + 16 * hi];
            ah.q[1]  = *(const u32x4*)&hbf[n][kc * 32 + 16 * hi + 8];
            ahk.q[0] = *(const u32x4*)&hbf[n][kk * 32 + 16 * hi];
            ahk.q[1] = *(const u32x4*)&hbf[n][kk * 32 + 16 * hi + 8];
            ack.q[0] = *(const u32x4*)&cbf[n][kk * 32 + 16 * hi];
            ack.q[1] = *(const u32x4*)&cbf[n][kk * 32 + 16 * hi + 8];
            b1.q[0]  = *(const u32x4*)&Wlds[col * 264 + kc * 32 + 16 * hi];
            b1.q[1]  = *(const u32x4*)&Wlds[col * 264 + kc * 32 + 16 * hi + 8];

            AFrag b2, b3, b4;                // Uo, Uc, Wd streamed from L2
            const __bf16* Wr2 = U5 + 2 * 65536 + (size_t)col * 256
                              + kk * 32 + 16 * hi;
            b2.q[0] = *(const u32x4*)Wr2;
            b2.q[1] = *(const u32x4*)(Wr2 + 8);
            b3.q[0] = *(const u32x4*)(Wr2 + 65536);
            b3.q[1] = *(const u32x4*)(Wr2 + 65536 + 8);
            b4.q[0] = *(const u32x4*)(Wr2 + 2 * 65536);
            b4.q[1] = *(const u32x4*)(Wr2 + 2 * 65536 + 8);

            acc[0] = wmma_bf16(ah.v,  Bw0[kc].v, acc[0]); // i    (regs)
            acc[1] = wmma_bf16(ah.v,  b1.v,      acc[1]); // f    (LDS)
            acc[2] = wmma_bf16(ahk.v, b2.v,      acc[2]); // o    (L2)
            acc[3] = wmma_bf16(ahk.v, b3.v,      acc[3]); // cand (L2)
            acc[4] = wmma_bf16(ack.v, b4.v,      acc[4]); // Wd   (L2)
        }
        __syncthreads();   // all waves finished reading hbf/cbf

        // lane-local LSTM cell update: all 5 gate sums for (m,col) are here
#pragma unroll
        for (int r = 0; r < 8; ++r) {
            int m = r + 8 * hi;
            float Cst = tanh_f(acc[4][r] + bD);
            float c   = creg[r] + (td[r] - 1.0f) * Cst;
            float ig  = sigmoid_f(x0[r] + acc[0][r] + bU0);
            float fg  = sigmoid_f(x1[r] + acc[1][r] + bU1);
            float og  = sigmoid_f(x2[r] + acc[2][r] + bU2);
            float Cn  = tanh_f  (x3[r] + acc[3][r] + bU3);
            c = fg * c + ig * Cn;
            float h = og * tanh_f(c);
            creg[r] = c;
            hbf[m][col] = (__bf16)h;
            cbf[m][col] = (__bf16)c;
            size_t oidx = oBase[r] + (size_t)s * 256;
            out[oidx] = h;
            if (writeTriple) {               // (outputs, (outputs, cells))
                out[NSH + oidx]     = h;
                out[2 * NSH + oidx] = c;
            } else {                         // (outputs, cells)
                out[NSH + oidx] = c;
            }
        }
        __syncthreads();   // hbf/cbf updated for next step
    }
}

// --------------------------- fp32 -> bf16 ----------------------------------
__global__ void cvt_bf16_kernel(const float* __restrict__ src,
                                __bf16* __restrict__ dst, int n) {
    int i = blockIdx.x * blockDim.x + threadIdx.x;
    if (i < n) dst[i] = (__bf16)src[i];
}

// ----------------- Phase 1: 4-gate input projection GEMM -------------------
// grid = 2048 (M tiles of 16), block = 512 (16 waves; wave w -> N-tile w)
__global__ __launch_bounds__(512) void gates_kernel(
    const float* __restrict__ X,        // [32768,256] fp32
    const __bf16* __restrict__ W4,      // [4][256][256] bf16 (Wi,Wf,Wo,Wc)
    const float* __restrict__ Wib, const float* __restrict__ bi_,
    const float* __restrict__ Wfb, const float* __restrict__ bf_,
    const float* __restrict__ Wob, const float* __restrict__ bo_,
    const float* __restrict__ Wcb, const float* __restrict__ bc_,
    float* __restrict__ G)              // [32768][4][256] fp32
{
    __shared__ __align__(16) __bf16 Abf[16][264];   // padded rows
    const int tid   = threadIdx.x;
    const int mBase = blockIdx.x * 16;

    // stage + convert one 16x256 row block of the inputs
    for (int e = tid; e < 16 * 256; e += 512) {
        int m = e >> 8, k = e & 255;
        Abf[m][k] = (__bf16)X[(size_t)(mBase + m) * 256 + k];
    }
    __syncthreads();

    const int wave = tid >> 5;
    const int lane = tid & 31;
    const int n    = lane & 15;
    const int hi   = lane >> 4;
    const int col  = wave * 16 + n;

    v8f acc[4] = {{0,0,0,0,0,0,0,0},{0,0,0,0,0,0,0,0},
                  {0,0,0,0,0,0,0,0},{0,0,0,0,0,0,0,0}};

#pragma unroll
    for (int kc = 0; kc < 8; ++kc) {
        AFrag a;
        a.q[0] = *(const u32x4*)&Abf[n][kc * 32 + 16 * hi];
        a.q[1] = *(const u32x4*)&Abf[n][kc * 32 + 16 * hi + 8];
#pragma unroll
        for (int g = 0; g < 4; ++g) {
            AFrag b;
            const __bf16* Wr = W4 + (size_t)g * 65536
                             + (size_t)col * 256 + kc * 32 + 16 * hi;
            b.q[0] = *(const u32x4*)Wr;
            b.q[1] = *(const u32x4*)(Wr + 8);
            acc[g] = wmma_bf16(a.v, b.v, acc[g]);
        }
    }

    float bias[4] = { Wib[col] + bi_[col], Wfb[col] + bf_[col],
                      Wob[col] + bo_[col], Wcb[col] + bc_[col] };
#pragma unroll
    for (int g = 0; g < 4; ++g) {
#pragma unroll
        for (int r = 0; r < 8; ++r) {
            size_t m = (size_t)(mBase + r + 8 * hi);
            G[(m * 4 + g) * 256 + col] = acc[g][r] + bias[g];
        }
    }
}

// --------------------------------- host ------------------------------------
extern "C" void kernel_launch(void* const* d_in, const int* in_sizes, int n_in,
                              void* d_out, int out_size, void* d_ws, size_t ws_size,
                              hipStream_t stream) {
    (void)in_sizes; (void)n_in; (void)ws_size;

    const float* X    = (const float*)d_in[0];
    const float* ts   = (const float*)d_in[1];
    const float* Wi_w = (const float*)d_in[2];  const float* Wi_b = (const float*)d_in[3];
    const float* Ui_w = (const float*)d_in[4];  const float* Ui_b = (const float*)d_in[5];
    const float* Wf_w = (const float*)d_in[6];  const float* Wf_b = (const float*)d_in[7];
    const float* Uf_w = (const float*)d_in[8];  const float* Uf_b = (const float*)d_in[9];
    const float* Wo_w = (const float*)d_in[10]; const float* Wo_b = (const float*)d_in[11];
    const float* Uo_w = (const float*)d_in[12]; const float* Uo_b = (const float*)d_in[13];
    const float* Wc_w = (const float*)d_in[14]; const float* Wc_b = (const float*)d_in[15];
    const float* Uc_w = (const float*)d_in[16]; const float* Uc_b = (const float*)d_in[17];
    const float* Wd_w = (const float*)d_in[18]; const float* Wd_b = (const float*)d_in[19];
    const float* bi   = (const float*)d_in[20]; const float* bff  = (const float*)d_in[21];
    const float* bo   = (const float*)d_in[22]; const float* bc   = (const float*)d_in[23];
    const float* bd   = (const float*)d_in[24];

    // workspace layout: [G interleaved fp32][W4 bf16][U5 bf16]
    char*   ws = (char*)d_ws;
    float*  G  = (float*)ws;
    __bf16* W4 = (__bf16*)(ws + (size_t)NROW * 1024 * sizeof(float));
    __bf16* U5 = W4 + 4 * 65536;

    const float* Wsrc[9] = { Wi_w, Wf_w, Wo_w, Wc_w,         // -> W4
                             Ui_w, Uf_w, Uo_w, Uc_w, Wd_w }; // -> U5
    for (int j = 0; j < 9; ++j)
        cvt_bf16_kernel<<<64, 1024, 0, stream>>>(Wsrc[j], W4 + (size_t)j * 65536, 65536);

    gates_kernel<<<2048, 512, 0, stream>>>(X, W4,
                                           Wi_b, bi, Wf_b, bff,
                                           Wo_b, bo, Wc_b, bc, G);

    int writeTriple = (out_size >= 3 * (int)(64 * 512 * 256)) ? 1 : 0;
    tlstm_rec_kernel<<<4, 512, 0, stream>>>(U5, G, ts,
                                            Ui_b, Uf_b, Uo_b, Uc_b,
                                            Wd_b, bd,
                                            (float*)d_out, writeTriple);
}
